// Model_24653112279778
// MI455X (gfx1250) — matrix operand order repaired and checked by reading, not compiled, not run on hardware
//
#include <hip/hip_runtime.h>
#include <hip/hip_bf16.h>

typedef __attribute__((ext_vector_type(16))) _Float16 v16h;
typedef __attribute__((ext_vector_type(8)))  _Float16 v8h;
typedef __attribute__((ext_vector_type(8)))  float    v8f;

__device__ __forceinline__ v16h cat8(v8h lo, v8h hi) {
  return __builtin_shufflevector(lo, hi, 0,1,2,3,4,5,6,7,8,9,10,11,12,13,14,15);
}
__device__ __forceinline__ v8f wmma_f16(v16h a, v16h b, v8f c) {
  return __builtin_amdgcn_wmma_f32_16x16x32_f16(false, a, false, b, (short)0, c, false, false);
}
__device__ __forceinline__ float sigm(float x) { return 1.f / (1.f + __expf(-x)); }

// ---------------- prep kernels ----------------

// weight-normed linear: WnH[r][d] = f16(g[r]*V[r][d]/||V[r]||), [32][128]
__global__ void k_prep_wn(const float* __restrict__ V, const float* __restrict__ g,
                          _Float16* __restrict__ WnH) {
  int r = threadIdx.x;
  if (r < 32) {
    float s = 0.f;
    for (int d = 0; d < 128; ++d) { float v = V[r*128 + d]; s += v*v; }
    float inv = g[r] / sqrtf(s);
    for (int d = 0; d < 128; ++d) WnH[r*128 + d] = (_Float16)(V[r*128 + d] * inv);
  }
}

// convert GRU weights to f16 (same [192][64] layout; rows are B-operand columns)
__global__ void k_cvt_wts(const float* __restrict__ Whh0, const float* __restrict__ Wih1,
                          _Float16* __restrict__ Whh0H, _Float16* __restrict__ Wih1H) {
  int idx = blockIdx.x * 256 + threadIdx.x;           // 48*256 == 12288 exactly
  Whh0H[idx] = (_Float16)Whh0[idx];
  Wih1H[idx] = (_Float16)Wih1[idx];
}

// GtH[j][i] = f16(G[i][j])
__global__ void k_cvt_graph(const float* __restrict__ G, _Float16* __restrict__ GtH) {
  int idx = blockIdx.x * 256 + threadIdx.x;           // 4096*256 == 1M exactly
  int i = idx >> 10, j = idx & 1023;
  GtH[(size_t)j * 1024 + i] = (_Float16)G[idx];
}

// gh1c[m][c] = H_init[1][m] . Whh1[c] + bhh1[c]   (constant over steps/batch)
__global__ void k_gh1c(const float* __restrict__ H1, const float* __restrict__ Whh1,
                       const float* __restrict__ bhh1, float* __restrict__ gh1c) {
  int idx = blockIdx.x * 256 + threadIdx.x;           // 768*256 == 196608 exactly
  int m = idx / 192, c = idx % 192;
  float s = bhh1[c];
  for (int k = 0; k < 64; ++k) s += H1[m*64 + k] * Whh1[c*64 + k];
  gh1c[(size_t)m * 192 + c] = s;
}

// hstate[b][m][k] = H_init[0][m][k]
__global__ void k_hinit(const float* __restrict__ H0, float* __restrict__ hstate) {
  size_t idx = (size_t)blockIdx.x * 256 + threadIdx.x; // 8192*256 == 2M exactly
  hstate[idx] = H0[idx & 65535];
}

// ---------------- GRU GEMM helper ----------------
// Gm[64][192] = Ah[64][64](f16) @ WH^T, WH is [192][64] f16.
// Tile map: rt = w>>1 (wave-constant -> A-frags loaded once), nt = (w&1)*6+q.
__device__ __forceinline__ void gru_gemm(const _Float16* __restrict__ Ah /*stride 72*/,
                                         const _Float16* __restrict__ WH,
                                         float* __restrict__ Gm /*stride 197*/,
                                         int w, int ln, int hi8, int hi16) {
  const int rt = w >> 1;
  const int r  = rt * 16 + ln;
  const _Float16* ar = Ah + (size_t)r * 72;
  const v16h a0 = cat8(*(const v8h*)(ar + hi8),      *(const v8h*)(ar + hi8 + 16));
  const v16h a1 = cat8(*(const v8h*)(ar + 32 + hi8), *(const v8h*)(ar + 32 + hi8 + 16));
  #pragma unroll
  for (int q = 0; q < 6; ++q) {
    const int nt = (w & 1) * 6 + q;
    const int c  = nt * 16 + ln;
    const _Float16* br = WH + (size_t)c * 64;
    v16h b0 = cat8(*(const v8h*)(br + hi8),      *(const v8h*)(br + hi8 + 16));
    v16h b1 = cat8(*(const v8h*)(br + 32 + hi8), *(const v8h*)(br + 32 + hi8 + 16));
    v8f acc = {};
    acc = wmma_f16(a0, b0, acc);
    acc = wmma_f16(a1, b1, acc);
    float* gm = Gm + (size_t)(rt * 16 + hi8) * 197 + nt * 16 + ln;
    #pragma unroll
    for (int gq = 0; gq < 8; ++gq) gm[(size_t)gq * 197] = acc[gq];
  }
}

// ---------------- per-step fused GRU kernel ----------------
// block = 64 rows of the flattened [B*M] row space (all in one batch b).
__global__ __launch_bounds__(256) void k_gru(
    const float* __restrict__ x_input, const float* __restrict__ Wih0,
    const float* __restrict__ bih0, const float* __restrict__ bhh0,
    const float* __restrict__ bih1,
    const _Float16* __restrict__ Whh0H, const _Float16* __restrict__ Wih1H,
    const float* __restrict__ gh1c, const float* __restrict__ H1,
    float* __restrict__ hstate, _Float16* __restrict__ XpT, int t)
{
  __shared__ _Float16 Ah[64][72];     // f16 A-operand (h, then hn0)
  __shared__ float    Gm[64][197];    // f32 GEMM output [64][192] padded
  __shared__ float    Xrow[64];

  const int tid = threadIdx.x;
  const int r0  = blockIdx.x * 64;
  const int b   = r0 >> 10;
  const int m0  = r0 & 1023;
  const int L   = tid & 31;
  const int w   = tid >> 5;
  const int ln  = L & 15;
  const int hi8  = (L >> 4) * 8;
  const int hi16 = (L >> 4) * 16;

  for (int e = tid; e < 4096; e += 256) {
    int r = e >> 6, k = e & 63;
    Ah[r][k] = (_Float16)hstate[(size_t)(r0 + r) * 64 + k];
  }
  if (tid < 64) Xrow[tid] = x_input[((size_t)b * 32 + t) * 1024 + m0 + tid];
  __syncthreads();

  // GEMM1: Gm = h(f16) @ Whh0^T
  gru_gemm(&Ah[0][0], Whh0H, &Gm[0][0], w, ln, hi8, hi16);
  __syncthreads();

  // epilogue 1: layer-0 GRU gates -> hn0 (f32 math), write f16 to Ah and XpT
  for (int e = tid; e < 4096; e += 256) {
    int r = e & 63, hd = e >> 6;
    float x = Xrow[r];
    float gir = x*Wih0[hd]      + bih0[hd];
    float giz = x*Wih0[64+hd]   + bih0[64+hd];
    float gin = x*Wih0[128+hd]  + bih0[128+hd];
    float ghr = Gm[r][hd]       + bhh0[hd];
    float ghz = Gm[r][64+hd]    + bhh0[64+hd];
    float ghn = Gm[r][128+hd]   + bhh0[128+hd];
    float rg = sigm(gir + ghr), zg = sigm(giz + ghz);
    float nn = tanhf(gin + rg*ghn);
    float h  = hstate[(size_t)(r0 + r) * 64 + hd];
    float hn0 = (1.f - zg)*nn + zg*h;
    Ah[r][hd] = (_Float16)hn0;
    XpT[(((size_t)b*128 + hd) << 10) + m0 + r] = (_Float16)hn0;
  }
  __syncthreads();

  // GEMM2: Gm = hn0(f16) @ Wih1^T
  gru_gemm(&Ah[0][0], Wih1H, &Gm[0][0], w, ln, hi8, hi16);
  __syncthreads();

  // epilogue 2: layer-1 GRU gates -> hn1; update hstate, write XpT[64..127]
  for (int e = tid; e < 4096; e += 256) {
    int r = e & 63, hd = e >> 6;
    int m = m0 + r;
    const float* gg = gh1c + (size_t)m * 192;
    float gir = Gm[r][hd]      + bih1[hd];
    float giz = Gm[r][64+hd]   + bih1[64+hd];
    float gin = Gm[r][128+hd]  + bih1[128+hd];
    float rg = sigm(gir + gg[hd]);
    float zg = sigm(giz + gg[64+hd]);
    float nn = tanhf(gin + rg*gg[128+hd]);
    float h1 = H1[(size_t)m * 64 + hd];
    float hn1 = (1.f - zg)*nn + zg*h1;
    hstate[(size_t)(r0 + r) * 64 + hd] = hn1;
    XpT[(((size_t)b*128 + 64 + hd) << 10) + m] = (_Float16)hn1;
  }
}

// ---------------- per-step graph GEMM + head (fused, double-buffered) ----------
// block: b = blockIdx.y, j-tile of 128 = blockIdx.x. 8 waves, wave tile 32x64.
// LDS arena: 2 stages x (As 10240B + Bs 10240B) = 40960B; phase-2 Et (34816B)
// overlaps the stages (only live after the K-loop, barrier-guarded).
// Stage pointers are computed as runtime byte offsets (no LDS-pointer arrays:
// constant-initialized aggregates of addrspacecast'd LDS addresses break lld).
__global__ __launch_bounds__(256) void k_graph(
    const _Float16* __restrict__ GtH, const _Float16* __restrict__ XpT,
    const _Float16* __restrict__ WnH, const float* __restrict__ b_lin,
    const float* __restrict__ W_fr, const float* __restrict__ b_fr,
    float* __restrict__ out, int t)
{
  __shared__ __align__(16) char smem[40960];

  const int tid = threadIdx.x;
  const int jt = blockIdx.x, b = blockIdx.y;
  const int jbase = jt * 128;
  const int L = tid & 31, w = tid >> 5;
  const int ln = L & 15;
  const int hi8  = (L >> 4) * 8;
  const int hi16 = (L >> 4) * 16;
  const int wj = (w >> 1) * 32, wd = (w & 1) * 64;

  // per-thread cooperative-load addressing: rows rl and rl+64, 8-half chunk kc
  const int rl = tid >> 2, kc = (tid & 3) << 3;
  const _Float16* gp0 = GtH + (size_t)(jbase + rl) * 1024 + kc;
  const _Float16* gp1 = gp0 + (size_t)64 * 1024;
  const _Float16* bp0 = XpT + (((size_t)b * 128 + rl) << 10) + kc;
  const _Float16* bp1 = bp0 + (size_t)64 * 1024;
  const int sa0 = rl * 80 + kc * 2;              // byte offsets within a stage
  const int sa1 = (rl + 64) * 80 + kc * 2;
  const int sb0 = 10240 + sa0, sb1 = 10240 + sa1;

  v8f zero = {};
  v8f acc[2][4];
  #pragma unroll
  for (int i = 0; i < 2; ++i)
    #pragma unroll
    for (int j = 0; j < 4; ++j) acc[i][j] = zero;

  // prologue: tile 0 -> stage 0
  v8h ra0 = *(const v8h*)(gp0), ra1 = *(const v8h*)(gp1);
  v8h rb0 = *(const v8h*)(bp0), rb1 = *(const v8h*)(bp1);
  *(v8h*)(smem + sa0) = ra0;  *(v8h*)(smem + sa1) = ra1;
  *(v8h*)(smem + sb0) = rb0;  *(v8h*)(smem + sb1) = rb1;

  int stage = 0;
  for (int it = 0; it < 32; ++it) {
    __syncthreads();
    // issue next tile's global loads early (overlaps with WMMA below)
    if (it < 31) {
      int kk = (it + 1) * 32;
      ra0 = *(const v8h*)(gp0 + kk);  ra1 = *(const v8h*)(gp1 + kk);
      rb0 = *(const v8h*)(bp0 + kk);  rb1 = *(const v8h*)(bp1 + kk);
    }
    // compute from current stage
    const char* sb = smem + stage * 20480;
    const _Float16* As = (const _Float16*)(sb);
    const _Float16* Bs = (const _Float16*)(sb + 10240);
    v16h af[2], bf[4];
    #pragma unroll
    for (int i = 0; i < 2; ++i) {
      const _Float16* ar = As + (size_t)(wj + i*16 + ln) * 40;
      af[i] = cat8(*(const v8h*)(ar + hi8), *(const v8h*)(ar + hi8 + 16));
    }
    #pragma unroll
    for (int j = 0; j < 4; ++j) {
      const _Float16* br = Bs + (size_t)(wd + j*16 + ln) * 40;
      bf[j] = cat8(*(const v8h*)(br + hi8), *(const v8h*)(br + hi8 + 16));
    }
    #pragma unroll
    for (int i = 0; i < 2; ++i)
      #pragma unroll
      for (int j = 0; j < 4; ++j)
        acc[i][j] = wmma_f16(af[i], bf[j], acc[i][j]);
    // fill the other stage (no race: readers use `stage`)
    if (it < 31) {
      char* sn = smem + (stage ^ 1) * 20480;
      *(v8h*)(sn + sa0) = ra0;  *(v8h*)(sn + sa1) = ra1;
      *(v8h*)(sn + sb0) = rb0;  *(v8h*)(sn + sb1) = rb1;
    }
    stage ^= 1;
  }
  __syncthreads();   // all stage reads done before Et overwrites the arena

  // relu -> f16 LDS tile Et[128][136]
  _Float16* Et = (_Float16*)smem;
  #pragma unroll
  for (int i = 0; i < 2; ++i)
    #pragma unroll
    for (int j = 0; j < 4; ++j)
      #pragma unroll
      for (int gq = 0; gq < 8; ++gq) {
        float v = acc[i][j][gq];
        Et[(size_t)(wj + i*16 + hi8 + gq) * 136 + wd + j*16 + ln] =
            (_Float16)(v > 0.f ? v : 0.f);
      }
  __syncthreads();

  // phase 2: y = sigmoid(Et @ Wn^T + b_lin); out = sum_rd y*W_fr + b_fr
  const int rw = w * 16;
  v8f y[2]; y[0] = zero; y[1] = zero;
  #pragma unroll
  for (int ks = 0; ks < 128; ks += 32) {
    const _Float16* ar = Et + (size_t)(rw + ln) * 136 + ks;
    v16h a = cat8(*(const v8h*)(ar + hi8), *(const v8h*)(ar + hi8 + 16));
    #pragma unroll
    for (int j = 0; j < 2; ++j) {
      const _Float16* br = WnH + (size_t)(j*16 + ln) * 128 + ks;
      v16h bb = cat8(*(const v8h*)(br + hi8), *(const v8h*)(br + hi8 + 16));
      y[j] = wmma_f16(a, bb, y[j]);
    }
  }
  float tot[8];
  #pragma unroll
  for (int gq = 0; gq < 8; ++gq) tot[gq] = 0.f;
  #pragma unroll
  for (int j = 0; j < 2; ++j) {
    int rd = j*16 + ln;
    float bl = b_lin[rd];
    #pragma unroll
    for (int gq = 0; gq < 8; ++gq) {
      int row = jbase + rw + hi8 + gq;
      float v = y[j][gq] + bl;
      float s = 1.f / (1.f + __expf(-v));
      float p = s * W_fr[(size_t)row * 32 + rd];
      p += __shfl_xor(p, 1, 32);
      p += __shfl_xor(p, 2, 32);
      p += __shfl_xor(p, 4, 32);
      p += __shfl_xor(p, 8, 32);
      tot[gq] += p;
    }
  }
  if (ln == 0) {
    #pragma unroll
    for (int gq = 0; gq < 8; ++gq) {
      int row = jbase + rw + hi8 + gq;
      out[((size_t)b * 32 + t) * 1024 + row] = tot[gq] + b_fr[row];
    }
  }
}

// ---------------- launch ----------------
extern "C" void kernel_launch(void* const* d_in, const int* in_sizes, int n_in,
                              void* d_out, int out_size, void* d_ws, size_t ws_size,
                              hipStream_t stream)
{
  const float* x_input = (const float*)d_in[0];
  const float* G       = (const float*)d_in[1];
  const float* H_init  = (const float*)d_in[2];
  const float* Wih0    = (const float*)d_in[3];
  const float* Whh0    = (const float*)d_in[4];
  const float* bih0    = (const float*)d_in[5];
  const float* bhh0    = (const float*)d_in[6];
  const float* Wih1    = (const float*)d_in[7];
  const float* Whh1    = (const float*)d_in[8];
  const float* bih1    = (const float*)d_in[9];
  const float* bhh1    = (const float*)d_in[10];
  const float* V       = (const float*)d_in[11];
  const float* gvec    = (const float*)d_in[12];
  const float* b_lin   = (const float*)d_in[13];
  const float* W_fr    = (const float*)d_in[14];
  const float* b_fr    = (const float*)d_in[15];
  float* out = (float*)d_out;

  char* ws = (char*)d_ws;
  size_t off = 0;
  _Float16* GtH   = (_Float16*)(ws + off); off += (size_t)1024 * 1024 * 2;
  _Float16* Whh0H = (_Float16*)(ws + off); off += (size_t)192 * 64 * 2;
  _Float16* Wih1H = (_Float16*)(ws + off); off += (size_t)192 * 64 * 2;
  _Float16* WnH   = (_Float16*)(ws + off); off += (size_t)32 * 128 * 2;
  float*    gh1c  = (float*)(ws + off);    off += (size_t)1024 * 192 * 4;
  float*    hstate= (float*)(ws + off);    off += (size_t)32768 * 64 * 4;
  _Float16* XpT   = (_Float16*)(ws + off); off += (size_t)32 * 128 * 1024 * 2;

  k_prep_wn  <<<1,    32, 0, stream>>>(V, gvec, WnH);
  k_cvt_wts  <<<48,  256, 0, stream>>>(Whh0, Wih1, Whh0H, Wih1H);
  k_cvt_graph<<<4096,256, 0, stream>>>(G, GtH);
  k_gh1c     <<<768, 256, 0, stream>>>(H_init + 1024*64, Whh1, bhh1, gh1c);
  k_hinit    <<<8192,256, 0, stream>>>(H_init, hstate);

  for (int t = 0; t < 32; ++t) {
    k_gru  <<<512, 256, 0, stream>>>(x_input, Wih0, bih0, bhh0, bih1,
                                     Whh0H, Wih1H, gh1c, H_init + 1024*64,
                                     hstate, XpT, t);
    k_graph<<<dim3(8, 32), 256, 0, stream>>>(GtH, XpT, WnH, b_lin, W_fr, b_fr, out, t);
  }
}
